// ScaledDotProductAttention_26336739459726
// MI455X (gfx1250) — matrix operand order repaired and checked by reading, not compiled, not run on hardware
//
#include <hip/hip_runtime.h>

typedef __attribute__((ext_vector_type(16))) __bf16 v16bf;
typedef __attribute__((ext_vector_type(8)))  __bf16 v8bf;
typedef __attribute__((ext_vector_type(8)))  float  v8f;

#define B_ 4
#define S_ 4096
#define D_ 64
#define WAVES_PER_BLOCK 8      // 2 query tiles x 4 key splits
#define NSPLIT 4
#define KT2 64                 // keys per iteration (2 fused 32-key sub-tiles)
#define ITPS (S_ / NSPLIT / KT2)  // 16 iterations per split

__device__ inline float fast_exp2(float x) {
#if __has_builtin(__builtin_amdgcn_exp2f)
  return __builtin_amdgcn_exp2f(x);
#else
  return exp2f(x);
#endif
}

__device__ inline unsigned pack_bf16x2(float a, float b) {
  unsigned short ua = __builtin_bit_cast(unsigned short, (__bf16)a);
  unsigned short ub = __builtin_bit_cast(unsigned short, (__bf16)b);
  return ((unsigned)ub << 16) | ua;
}

// max-reduce across each 16-lane group on the VALU (v_permlane16_b32),
// avoiding LDS round-trips + dscnt waits of ds_bpermute.
__device__ inline float row16_max(float x) {
#if __has_builtin(__builtin_amdgcn_permlane16)
  unsigned u, t;
  u = __builtin_bit_cast(unsigned, x);                       // xor 1
  t = __builtin_amdgcn_permlane16(u, u, 0x67452301u, 0xEFCDAB89u, false, false);
  x = fmaxf(x, __builtin_bit_cast(float, t));
  u = __builtin_bit_cast(unsigned, x);                       // xor 2
  t = __builtin_amdgcn_permlane16(u, u, 0x54761032u, 0xDCFE98BAu, false, false);
  x = fmaxf(x, __builtin_bit_cast(float, t));
  u = __builtin_bit_cast(unsigned, x);                       // xor 4
  t = __builtin_amdgcn_permlane16(u, u, 0x32107654u, 0xBA98FEDCu, false, false);
  x = fmaxf(x, __builtin_bit_cast(float, t));
  u = __builtin_bit_cast(unsigned, x);                       // xor 8
  t = __builtin_amdgcn_permlane16(u, u, 0xFEDCBA98u, 0x76543210u, false, false);
  x = fmaxf(x, __builtin_bit_cast(float, t));
  return x;
#else
  #pragma unroll
  for (int dx = 1; dx < 16; dx <<= 1)
    x = fmaxf(x, __shfl_xor(x, dx, 32));
  return x;
#endif
}

// ---- pass 1: fp32 -> bf16 convert. K straight; V transposed AND key-permuted
// within each 32-key group: pos(w) = (w<16) ? 2w : 2(w-16)+1, so that the PV
// contraction order k' = 2*col + half matches a packed (p0,p1) dword store.
__global__ __launch_bounds__(256) void cvt_kernel(
    const float* __restrict__ k, const float* __restrict__ v,
    __bf16* __restrict__ kb, __bf16* __restrict__ vt) {
  size_t i = (size_t)blockIdx.x * 256 + threadIdx.x;   // over B*S*D
  kb[i] = (__bf16)k[i];
  int d  = (int)(i & 63);
  size_t bs = i >> 6;
  int s  = (int)(bs & (S_ - 1));
  int b  = (int)(bs >> 12);
  int w  = s & 31;
  int g  = s >> 5;
  int pos = (w < 16) ? (w << 1) : (((w - 16) << 1) | 1);
  int sp  = (g << 5) | pos;
  vt[((size_t)(b * D_ + d) << 12) + sp] = (__bf16)v[i]; // Vt[b][d][perm(s)]
}

// ------- pass 2: split-K flash attention, 1 wave = 16 q-rows x 1024 keys -----
__global__ __launch_bounds__(256) void attn_kernel(
    const float* __restrict__ q, const float* __restrict__ mask,
    const __bf16* __restrict__ kb, const __bf16* __restrict__ vt,
    float* __restrict__ out) {
  __shared__ __align__(64) unsigned plu[WAVES_PER_BLOCK][16][32]; // P transpose
  __shared__ float OL[WAVES_PER_BLOCK][16][68];                   // O + L
  __shared__ float Mv[WAVES_PER_BLOCK][16];                       // max

  const int lane  = threadIdx.x & 31;
  const int wave  = threadIdx.x >> 5;
  const int qgrp  = wave >> 2;                   // 0..1 : q-tile within block
  const int split = wave & 3;                    // 0..3 : key split
  const int gt    = blockIdx.x * 2 + qgrp;       // global q-tile 0..1023
  const int b     = gt >> 8;
  const int q0    = (gt & 255) << 4;
  const int col   = lane & 15;
  const int hi    = lane >> 4;

  const float C = 0.125f * 1.44269504088896340736f;  // (1/sqrt(64))*log2(e)

  // ---- Q A-fragments, pre-scaled by C (row = lane%16, contiguous k = 16*hi + s) --
  v16bf aq[2];
  {
    const float* qrow = q + (((size_t)b * S_ + q0 + col) * D_);
    #pragma unroll
    for (int blk = 0; blk < 2; ++blk) {
      #pragma unroll
      for (int h = 0; h < 2; ++h) {
        int r0 = blk * 32 + hi * 16 + h * 8;
        const float4* p = (const float4*)(qrow + r0);
        float4 f0 = p[0], f1 = p[1];
        int o = h * 8;
        aq[blk][o+0] = (__bf16)(C*f0.x); aq[blk][o+1] = (__bf16)(C*f0.y);
        aq[blk][o+2] = (__bf16)(C*f0.z); aq[blk][o+3] = (__bf16)(C*f0.w);
        aq[blk][o+4] = (__bf16)(C*f1.x); aq[blk][o+5] = (__bf16)(C*f1.y);
        aq[blk][o+6] = (__bf16)(C*f1.z); aq[blk][o+7] = (__bf16)(C*f1.w);
      }
    }
  }

  // ones B-fragment: O_L = P * ones accumulates the softmax denominator
  v16bf vones;
  #pragma unroll
  for (int i2 = 0; i2 < 16; ++i2) vones[i2] = (__bf16)1.0f;

  float Mrow[8];
  const v8f vzero = {0.f,0.f,0.f,0.f,0.f,0.f,0.f,0.f};
  v8f oacc[4], oaccL;
  #pragma unroll
  for (int r = 0; r < 8; ++r) Mrow[r] = -__builtin_inff();
  #pragma unroll
  for (int t = 0; t < 4; ++t) oacc[t] = vzero;
  oaccL = vzero;

  const float* mbase = mask + (((size_t)b * S_ + q0 + hi * 8) * S_) + col;
  const __bf16* kbb  = kb + ((size_t)b * S_ * D_);
  const __bf16* vtb  = vt + ((size_t)b * D_ * S_);

  const int jend = split * ITPS + ITPS;
  for (int j = split * ITPS; j < jend; ++j) {
    const int k0 = j * KT2;
    if (j + 1 < jend)
      __builtin_prefetch(mbase + k0 + KT2, 0, 0);  // stream next mask tiles

    // ---- scores: S16x64 = (C*Q)(16x64) . K^T, 4 column tiles of 16 ----
    v8f sc[4] = { vzero, vzero, vzero, vzero };
    #pragma unroll
    for (int cN = 0; cN < 4; ++cN) {
      const __bf16* krow = kbb + ((size_t)(k0 + cN * 16 + col) * D_) + hi * 16;
      #pragma unroll
      for (int blk = 0; blk < 2; ++blk) {
        v16bf bk = *(const v16bf*)(krow + blk * 32);
        sc[cN] = __builtin_amdgcn_wmma_f32_16x16x32_bf16(
            false, aq[blk], false, bk, (short)0, sc[cN], false, false);
      }
    }

    // ---- multiplicative mask + online softmax, one rescale per 64 keys ----
    #pragma unroll
    for (int r = 0; r < 8; ++r) {
      float s0 = sc[0][r] * mbase[(size_t)r * S_ + k0];
      float s1 = sc[1][r] * mbase[(size_t)r * S_ + k0 + 16];
      float s2 = sc[2][r] * mbase[(size_t)r * S_ + k0 + 32];
      float s3 = sc[3][r] * mbase[(size_t)r * S_ + k0 + 48];
      float tmax = row16_max(fmaxf(fmaxf(s0, s1), fmaxf(s2, s3)));
      float Mn = fmaxf(Mrow[r], tmax);
      float f  = fast_exp2(Mrow[r] - Mn);
      float p0 = fast_exp2(s0 - Mn);
      float p1 = fast_exp2(s1 - Mn);
      float p2 = fast_exp2(s2 - Mn);
      float p3 = fast_exp2(s3 - Mn);
      Mrow[r] = Mn;
      #pragma unroll
      for (int t = 0; t < 4; ++t) oacc[t][r] *= f;
      oaccL[r] *= f;
      int rowm = r + hi * 8;
      plu[wave][rowm][col]      = pack_bf16x2(p0, p1);  // sub-tile 0
      plu[wave][rowm][col + 16] = pack_bf16x2(p2, p3);  // sub-tile 1
    }

    // ---- PV: two 32-key sub-tiles; denominator via ones-column WMMA ----
    #pragma unroll
    for (int sub = 0; sub < 2; ++sub) {
      v16bf ap;
      {
        const __bf16* prow = (const __bf16*)&plu[wave][col][0] + sub * 32;
        v8bf a0 = *(const v8bf*)(prow + hi * 16);
        v8bf a1 = *(const v8bf*)(prow + hi * 16 + 8);
        #pragma unroll
        for (int i2 = 0; i2 < 8; ++i2) { ap[i2] = a0[i2]; ap[8 + i2] = a1[i2]; }
      }
      #pragma unroll
      for (int t = 0; t < 4; ++t) {
        const __bf16* vrow = vtb + ((size_t)(t * 16 + col) * S_)
                           + k0 + sub * 32 + hi * 16;
        v16bf bv = *(const v16bf*)vrow;   // key-permuted layout matches k'
        oacc[t] = __builtin_amdgcn_wmma_f32_16x16x32_bf16(
            false, ap, false, bv, (short)0, oacc[t], false, false);
      }
      oaccL = __builtin_amdgcn_wmma_f32_16x16x32_bf16(
          false, ap, false, vones, (short)0, oaccL, false, false);
    }
  }

  // ---- publish split partials to LDS ----
  #pragma unroll
  for (int r = 0; r < 8; ++r) {
    int rowm = r + hi * 8;
    #pragma unroll
    for (int t = 0; t < 4; ++t) OL[wave][rowm][t * 16 + col] = oacc[t][r];
    if (col == 0) {
      OL[wave][rowm][64] = oaccL[r];
      Mv[wave][rowm]     = Mrow[r];
    }
  }
  __syncthreads();

  // ---- merge 4 splits (one wave per q-tile does the reduction) ----
  if (split == 0) {
    const int wb = qgrp * 4;
    for (int m = 0; m < 16; ++m) {
      float m0 = Mv[wb][m], m1 = Mv[wb + 1][m];
      float m2 = Mv[wb + 2][m], m3 = Mv[wb + 3][m];
      float Ms = fmaxf(fmaxf(m0, m1), fmaxf(m2, m3));
      float w0 = fast_exp2(m0 - Ms), w1 = fast_exp2(m1 - Ms);
      float w2 = fast_exp2(m2 - Ms), w3 = fast_exp2(m3 - Ms);
      float a0 = w0 * OL[wb][m][lane]      + w1 * OL[wb + 1][m][lane]
               + w2 * OL[wb + 2][m][lane]  + w3 * OL[wb + 3][m][lane];
      float a1 = w0 * OL[wb][m][lane + 32]     + w1 * OL[wb + 1][m][lane + 32]
               + w2 * OL[wb + 2][m][lane + 32] + w3 * OL[wb + 3][m][lane + 32];
      float Lc = w0 * OL[wb][m][64]     + w1 * OL[wb + 1][m][64]
               + w2 * OL[wb + 2][m][64] + w3 * OL[wb + 3][m][64];
      float inv = 1.0f / Lc;
      float* orow = out + (((size_t)b * S_ + q0 + m) * D_);
      orow[lane]      = a0 * inv;
      orow[lane + 32] = a1 * inv;
    }
  }
}

extern "C" void kernel_launch(void* const* d_in, const int* in_sizes, int n_in,
                              void* d_out, int out_size, void* d_ws, size_t ws_size,
                              hipStream_t stream) {
  const float* q    = (const float*)d_in[0];
  const float* k    = (const float*)d_in[1];
  const float* v    = (const float*)d_in[2];
  const float* mask = (const float*)d_in[3];
  float* out = (float*)d_out;

  __bf16* kb = (__bf16*)d_ws;                         // 2 MB
  __bf16* vt = kb + (size_t)B_ * S_ * D_;             // 2 MB

  int n = B_ * S_ * D_;                               // 1M elements
  cvt_kernel<<<n / 256, 256, 0, stream>>>(k, v, kb, vt);

  int blocks = (B_ * (S_ / 16)) / 2;                  // 1024 q-tiles, 2/block
  attn_kernel<<<blocks, 256, 0, stream>>>(q, mask, kb, vt, out);
}
